// GNN_32152125178567
// MI455X (gfx1250) — matrix operand order repaired and checked by reading, not compiled, not run on hardware
//
#include <hip/hip_runtime.h>
#include <hip/hip_bf16.h>
#include <climits>

typedef __attribute__((ext_vector_type(16))) _Float16 v16h;
typedef __attribute__((ext_vector_type(8)))  _Float16 v8h;
typedef __attribute__((ext_vector_type(4)))  _Float16 v4h;
typedef __attribute__((ext_vector_type(8)))  float    v8f;
typedef __attribute__((ext_vector_type(4)))  float    v4f;

#define HCH 128

// ---------- helpers ----------
__device__ __forceinline__ int fenc(float f) {
    int i = __float_as_int(f);
    return (i >= 0) ? i : (i ^ 0x7FFFFFFF);
}
__device__ __forceinline__ float fdec(int k) {
    return __int_as_float((k >= 0) ? k : (k ^ 0x7FFFFFFF));
}

// ---------- generic fills ----------
__global__ void fill_f32(float* __restrict__ p, float v, long long n) {
    long long i = (long long)blockIdx.x * blockDim.x + threadIdx.x;
    if (i < n) p[i] = v;
}
__global__ void fill_i32(int* __restrict__ p, int v, long long n) {
    long long i = (long long)blockIdx.x * blockDim.x + threadIdx.x;
    if (i < n) p[i] = v;
}

// ---------- weight f32 -> f16, TRANSPOSED with K padding ----------
// src: [srcRows, 128] row-major f32;  dst: [128, Kpad] row-major f16 (i.e. dst[c][k] = src[k][c])
__global__ void convert_padT(const float* __restrict__ src, _Float16* __restrict__ dst,
                             int srcRows, int Kpad) {
    int i = blockIdx.x * blockDim.x + threadIdx.x;
    if (i >= Kpad * HCH) return;
    int c = i / Kpad;
    int k = i - c * Kpad;
    dst[i] = (k < srcRows) ? (_Float16)src[(size_t)k * HCH + c] : (_Float16)0.0f;
}

// ---------- feature construction ----------
__global__ void edge_feat_kernel(const int* __restrict__ bond_type,
                                 const float* __restrict__ edge_cont,
                                 const int* __restrict__ bond_conj,
                                 const int* __restrict__ bond_arom,
                                 const int* __restrict__ dst,
                                 const float* __restrict__ bond_emb,   // [22,8]
                                 const float* __restrict__ bool_emb,   // [3,2]
                                 _Float16* __restrict__ eafull16,      // [(E+N),32]
                                 float* __restrict__ deg,              // [N]
                                 float* __restrict__ easum,            // [N,16]
                                 int E) {
    int e = blockIdx.x * blockDim.x + threadIdx.x;
    if (e >= E) return;
    float ea[13];
    const float* be = bond_emb + bond_type[e] * 8;
#pragma unroll
    for (int j = 0; j < 8; ++j) ea[j] = be[j];
    ea[8] = edge_cont[e];
    const float* bc = bool_emb + bond_conj[e] * 2;
    ea[9] = bc[0]; ea[10] = bc[1];
    const float* ba = bool_emb + bond_arom[e] * 2;
    ea[11] = ba[0]; ea[12] = ba[1];
    _Float16* row = eafull16 + (size_t)e * 32;
#pragma unroll
    for (int j = 0; j < 13; ++j) row[j] = (_Float16)ea[j];
#pragma unroll
    for (int j = 13; j < 32; ++j) row[j] = (_Float16)0.0f;
    int d = dst[e];
    atomicAdd(&deg[d], 1.0f);
#pragma unroll
    for (int j = 0; j < 13; ++j) atomicAdd(&easum[(size_t)d * 16 + j], ea[j]);
}

__global__ void node_feat_kernel(const int* __restrict__ atom_num,
                                 const int* __restrict__ atom_arom,
                                 const float* __restrict__ x_cont,     // [N,8]
                                 const float* __restrict__ atom_emb,   // [119,16]
                                 const float* __restrict__ bool_emb,   // [3,2]
                                 const float* __restrict__ deg,
                                 const float* __restrict__ easum,      // [N,16]
                                 _Float16* __restrict__ feat16,        // [N,32] (K=32 layout)
                                 _Float16* __restrict__ eafull16,      // rows E..E+N
                                 int N, int E) {
    int n = blockIdx.x * blockDim.x + threadIdx.x;
    if (n >= N) return;
    _Float16* row = feat16 + (size_t)n * 32;
    const float* ae = atom_emb + atom_num[n] * 16;
#pragma unroll
    for (int j = 0; j < 16; ++j) row[j] = (_Float16)ae[j];
    const float* xc = x_cont + (size_t)n * 8;
#pragma unroll
    for (int j = 0; j < 8; ++j) row[16 + j] = (_Float16)xc[j];
    const float* aa = bool_emb + atom_arom[n] * 2;
    row[24] = (_Float16)aa[0];
    row[25] = (_Float16)aa[1];
#pragma unroll
    for (int j = 26; j < 32; ++j) row[j] = (_Float16)0.0f;
    // self-loop edge feature = mean of incoming edge features
    float g = fmaxf(deg[n], 1.0f);
    _Float16* er = eafull16 + (size_t)(E + n) * 32;
#pragma unroll
    for (int j = 0; j < 13; ++j) er[j] = (_Float16)(easum[(size_t)n * 16 + j] / g);
#pragma unroll
    for (int j = 13; j < 32; ++j) er[j] = (_Float16)0.0f;
}

// ---------- WMMA GEMM: out[rows,128] = A16[rows,K] * W[K,128] (+ bias) ----------
// W is given TRANSPOSED: W16T[col][k], row pitch = K halves.
// block = 256 threads = 8 waves; block tile = 64 rows x 128 cols.
// wave w -> 16-col slice w, 4 row tiles (4 accumulators -> 4 WMMAs per K-step).
#define A_PITCH 40   // halves; 80B row stride, 16B aligned
__launch_bounds__(256)
__global__ void gemm_wmma(const _Float16* __restrict__ A16,
                          const _Float16* __restrict__ W16T,
                          const float* __restrict__ bias,      // [128] or nullptr
                          float* __restrict__ out,             // [rows,128]
                          int rows, int K) {
    __shared__ __align__(16) _Float16 As[64][A_PITCH];

    const int tid  = threadIdx.x;
    const int lane = tid & 31;
    const int wid  = tid >> 5;          // 0..7  -> N (column) tile
    const int l15  = lane & 15;
    const int hh   = lane >> 4;         // lane half
    const int rowbase = blockIdx.x * 64;
    const int col = wid * 16 + l15;

    v8f acc0 = {}, acc1 = {}, acc2 = {}, acc3 = {};

    const _Float16* wcol = W16T + (size_t)col * K;  // this lane's weight column

    for (int k0 = 0; k0 < K; k0 += 32) {
        // stage A tile: 64 rows x 32 halves; one b128 per thread
        {
            int r  = tid >> 2;          // 0..63
            int ck = (tid & 3) * 8;     // 0,8,16,24
            v8h av = {};
            int gr = rowbase + r;
            if (gr < rows)
                av = *(const v8h*)(A16 + (size_t)gr * K + k0 + ck);
            *(v8h*)&As[r][ck] = av;
        }
        __syncthreads();

        // B fragment straight from global (transposed weights, contiguous K):
        // b[i] = W[k0 + 16*hh + i][col]
        v8h blo = *(const v8h*)(wcol + k0 + hh * 16);
        v8h bhi = *(const v8h*)(wcol + k0 + hh * 16 + 8);
        v16h b = __builtin_shufflevector(blo, bhi, 0,1,2,3,4,5,6,7,8,9,10,11,12,13,14,15);

        // 4 row tiles: A fragment contiguous K: a[i] = A[row][k0 + 16*hh + i]
#pragma unroll
        for (int rt = 0; rt < 4; ++rt) {
            const _Float16* arow = &As[rt * 16 + l15][0];
            v8h alo = *(const v8h*)(arow + hh * 16);
            v8h ahi = *(const v8h*)(arow + hh * 16 + 8);
            v16h a = __builtin_shufflevector(alo, ahi, 0,1,2,3,4,5,6,7,8,9,10,11,12,13,14,15);
            v8f* accp = (rt == 0) ? &acc0 : (rt == 1) ? &acc1 : (rt == 2) ? &acc2 : &acc3;
            *accp = __builtin_amdgcn_wmma_f32_16x16x32_f16(
                false, a, false, b, (short)0, *accp, false, false);
        }
        __syncthreads();
    }

    float bv = bias ? bias[col] : 0.0f;
#pragma unroll
    for (int rt = 0; rt < 4; ++rt) {
        v8f acc = (rt == 0) ? acc0 : (rt == 1) ? acc1 : (rt == 2) ? acc2 : acc3;
#pragma unroll
        for (int r = 0; r < 8; ++r) {
            int row = rowbase + rt * 16 + r + 8 * hh;   // C/D: VGPR r -> M = r + 8*half
            if (row < rows) out[(size_t)row * HCH + col] = acc[r] + bv;
        }
    }
}

// ---------- edge pass A: message + leaky relu + attention logit + segment max ----------
// wave per edge; lane owns channels [4*lane, 4*lane+4) -> b128 loads
__launch_bounds__(256)
__global__ void passA_kernel(const int* __restrict__ src, const int* __restrict__ dst,
                             const float* __restrict__ xl, const float* __restrict__ xr,
                             const float* __restrict__ eh, const float* __restrict__ att,
                             float* __restrict__ logit, int* __restrict__ mx,
                             int E, int Etot) {
    int e = (int)(((long long)blockIdx.x * blockDim.x + threadIdx.x) >> 5);
    int lane = threadIdx.x & 31;
    if (e >= Etot) return;
    int s, d;
    if (e < E) { s = src[e]; d = dst[e]; } else { s = d = e - E; }
    int c = lane * 4;
    v4f vl = *(const v4f*)(xl + (size_t)s * HCH + c);
    v4f vr = *(const v4f*)(xr + (size_t)d * HCH + c);
    v4f ve = *(const v4f*)(eh + (size_t)e * HCH + c);
    v4f va = *(const v4f*)(att + c);
    float dot = 0.0f;
#pragma unroll
    for (int j = 0; j < 4; ++j) {
        float v = vl[j] + vr[j] + ve[j];
        v = (v > 0.0f) ? v : 0.2f * v;       // leaky_relu(0.2)
        dot += v * va[j];
    }
#pragma unroll
    for (int off = 16; off > 0; off >>= 1) dot += __shfl_xor(dot, off, 32);
    if (lane == 0) {
        logit[e] = dot;
        atomicMax(&mx[d], fenc(dot));
    }
}

// ---------- edge pass B: exp + segment sum ----------
__global__ void passB_kernel(const int* __restrict__ dst, const float* __restrict__ logit,
                             const int* __restrict__ mx, float* __restrict__ av,
                             float* __restrict__ denom, int E, int Etot) {
    int e = blockIdx.x * blockDim.x + threadIdx.x;
    if (e >= Etot) return;
    int d = (e < E) ? dst[e] : (e - E);
    float a = __expf(logit[e] - fdec(mx[d]));
    av[e] = a;
    atomicAdd(&denom[d], a);
}

// ---------- edge pass C: weighted aggregation ----------
__launch_bounds__(256)
__global__ void passC_kernel(const int* __restrict__ src, const int* __restrict__ dst,
                             const float* __restrict__ xl, const float* __restrict__ av,
                             const float* __restrict__ denom, float* __restrict__ agg,
                             int E, int Etot) {
    int e = (int)(((long long)blockIdx.x * blockDim.x + threadIdx.x) >> 5);
    int lane = threadIdx.x & 31;
    if (e >= Etot) return;
    int s, d;
    if (e < E) { s = src[e]; d = dst[e]; } else { s = d = e - E; }
    float alpha = av[e] / denom[d];
    int c = lane * 4;
    v4f vl = *(const v4f*)(xl + (size_t)s * HCH + c);
#pragma unroll
    for (int j = 0; j < 4; ++j)
        atomicAdd(&agg[(size_t)d * HCH + c + j], alpha * vl[j]);
}

// ---------- finalize: + bias, ReLU, write f32 + f16 mirror (4 elems/thread) ----------
__global__ void finalize_kernel(const float* __restrict__ agg, const float* __restrict__ bias,
                                float* __restrict__ hf, _Float16* __restrict__ feat16,
                                long long total) {
    long long i4 = ((long long)blockIdx.x * blockDim.x + threadIdx.x) * 4;
    if (i4 >= total) return;
    int c = (int)(i4 & (HCH - 1));
    v4f a = *(const v4f*)(agg + i4);
    v4f bv = *(const v4f*)(bias + c);
    v4f o; v4h o16;
#pragma unroll
    for (int j = 0; j < 4; ++j) {
        float v = fmaxf(a[j] + bv[j], 0.0f);
        o[j] = v;
        o16[j] = (_Float16)v;
    }
    *(v4f*)(hf + i4) = o;
    *(v4h*)(feat16 + i4) = o16;
}

// ---------- global max pool (post-ReLU values >= 0 -> uint atomicMax on f32 bits) ----------
__launch_bounds__(256)
__global__ void pool_kernel(const float* __restrict__ hf, const int* __restrict__ batch,
                            unsigned int* __restrict__ outp, int N) {
    int n = (int)(((long long)blockIdx.x * blockDim.x + threadIdx.x) >> 5);
    int lane = threadIdx.x & 31;
    if (n >= N) return;
    int g = batch[n];
    int c = lane * 4;
    v4f v = *(const v4f*)(hf + (size_t)n * HCH + c);
#pragma unroll
    for (int j = 0; j < 4; ++j)
        atomicMax(&outp[(size_t)g * HCH + c + j], __float_as_uint(v[j]));
}

// =====================================================================
extern "C" void kernel_launch(void* const* d_in, const int* in_sizes, int n_in,
                              void* d_out, int out_size, void* d_ws, size_t ws_size,
                              hipStream_t stream) {
    const int N = in_sizes[0];
    const int E = in_sizes[3];
    const int Etot = E + N;

    const int*   atom_num  = (const int*)d_in[0];
    const int*   atom_arom = (const int*)d_in[1];
    const float* x_cont    = (const float*)d_in[2];
    const int*   bond_type = (const int*)d_in[3];
    const float* edge_cont = (const float*)d_in[4];
    const int*   bond_conj = (const int*)d_in[5];
    const int*   bond_arom = (const int*)d_in[6];
    const int*   src       = (const int*)d_in[7];          // edge_index[0]
    const int*   dst       = ((const int*)d_in[7]) + E;    // edge_index[1]
    const int*   batch     = (const int*)d_in[8];
    const float* atom_emb  = (const float*)d_in[9];
    const float* bond_emb  = (const float*)d_in[10];
    const float* bool_emb  = (const float*)d_in[11];
    const float* Wl1 = (const float*)d_in[12]; const float* bl1 = (const float*)d_in[13];
    const float* Wr1 = (const float*)d_in[14]; const float* br1 = (const float*)d_in[15];
    const float* We1 = (const float*)d_in[16];
    const float* att1 = (const float*)d_in[17]; const float* bias1 = (const float*)d_in[18];
    const float* Wl2 = (const float*)d_in[19]; const float* bl2 = (const float*)d_in[20];
    const float* Wr2 = (const float*)d_in[21]; const float* br2 = (const float*)d_in[22];
    const float* We2 = (const float*)d_in[23];
    const float* att2 = (const float*)d_in[24]; const float* bias2 = (const float*)d_in[25];

    // ---- workspace carve-up ----
    char* p = (char*)d_ws;
    auto alloc = [&](size_t bytes) -> char* {
        char* r = p;
        p += (bytes + 255) & ~(size_t)255;
        return r;
    };
    _Float16* Wl1_16 = (_Float16*)alloc(32  * HCH * sizeof(_Float16));  // [128,32] transposed
    _Float16* Wr1_16 = (_Float16*)alloc(32  * HCH * sizeof(_Float16));
    _Float16* We1_16 = (_Float16*)alloc(32  * HCH * sizeof(_Float16));
    _Float16* Wl2_16 = (_Float16*)alloc(HCH * HCH * sizeof(_Float16));  // [128,128] transposed
    _Float16* Wr2_16 = (_Float16*)alloc(HCH * HCH * sizeof(_Float16));
    _Float16* We2_16 = (_Float16*)alloc(32  * HCH * sizeof(_Float16));
    _Float16* feat16   = (_Float16*)alloc((size_t)N * HCH * sizeof(_Float16));   // K=32 first, K=128 later
    _Float16* eafull16 = (_Float16*)alloc((size_t)Etot * 32 * sizeof(_Float16));
    float* eh    = (float*)alloc((size_t)Etot * HCH * sizeof(float));
    float* xl    = (float*)alloc((size_t)N * HCH * sizeof(float));
    float* xr    = (float*)alloc((size_t)N * HCH * sizeof(float));
    float* hf    = (float*)alloc((size_t)N * HCH * sizeof(float));
    float* agg   = (float*)alloc((size_t)N * HCH * sizeof(float));
    float* deg   = (float*)alloc((size_t)N * sizeof(float));
    float* easum = (float*)alloc((size_t)N * 16 * sizeof(float));
    int*   mx    = (int*)alloc((size_t)N * sizeof(int));
    float* denom = (float*)alloc((size_t)N * sizeof(float));
    float* logit = (float*)alloc((size_t)Etot * sizeof(float));
    float* av    = (float*)alloc((size_t)Etot * sizeof(float));
    (void)ws_size;

    const int T = 256;
    auto blk1 = [&](long long n) { return (unsigned)((n + T - 1) / T); };

    // ---- weights -> f16, transposed, K padded ----
    convert_padT<<<blk1(32 * HCH), T, 0, stream>>>(Wl1, Wl1_16, 26, 32);
    convert_padT<<<blk1(32 * HCH), T, 0, stream>>>(Wr1, Wr1_16, 26, 32);
    convert_padT<<<blk1(32 * HCH), T, 0, stream>>>(We1, We1_16, 13, 32);
    convert_padT<<<blk1(HCH * HCH), T, 0, stream>>>(Wl2, Wl2_16, HCH, HCH);
    convert_padT<<<blk1(HCH * HCH), T, 0, stream>>>(Wr2, Wr2_16, HCH, HCH);
    convert_padT<<<blk1(32 * HCH), T, 0, stream>>>(We2, We2_16, 13, 32);

    // ---- features + self-loop edge means ----
    fill_f32<<<blk1(N), T, 0, stream>>>(deg, 0.0f, N);
    fill_f32<<<blk1((long long)N * 16), T, 0, stream>>>(easum, 0.0f, (long long)N * 16);
    edge_feat_kernel<<<blk1(E), T, 0, stream>>>(bond_type, edge_cont, bond_conj, bond_arom,
                                                dst, bond_emb, bool_emb, eafull16, deg, easum, E);
    node_feat_kernel<<<blk1(N), T, 0, stream>>>(atom_num, atom_arom, x_cont, atom_emb, bool_emb,
                                                deg, easum, feat16, eafull16, N, E);

    const unsigned gemmN  = (unsigned)((N + 63) / 64);
    const unsigned gemmEt = (unsigned)((Etot + 63) / 64);
    const unsigned edgeW  = (unsigned)((Etot + 7) / 8);   // wave-per-edge, 8 waves/block

    auto run_layer = [&](const _Float16* feat, int K,
                         const _Float16* WlT, const float* bl,
                         const _Float16* WrT, const float* br,
                         const float* att, const float* bias) {
        gemm_wmma<<<gemmN, T, 0, stream>>>(feat, WlT, bl, xl, N, K);
        gemm_wmma<<<gemmN, T, 0, stream>>>(feat, WrT, br, xr, N, K);
        fill_i32<<<blk1(N), T, 0, stream>>>(mx, INT_MIN, N);
        fill_f32<<<blk1(N), T, 0, stream>>>(denom, 0.0f, N);
        fill_f32<<<blk1((long long)N * HCH), T, 0, stream>>>(agg, 0.0f, (long long)N * HCH);
        passA_kernel<<<edgeW, T, 0, stream>>>(src, dst, xl, xr, eh, att, logit, mx, E, Etot);
        passB_kernel<<<blk1(Etot), T, 0, stream>>>(dst, logit, mx, av, denom, E, Etot);
        passC_kernel<<<edgeW, T, 0, stream>>>(src, dst, xl, av, denom, agg, E, Etot);
        finalize_kernel<<<blk1(((long long)N * HCH) / 4), T, 0, stream>>>(agg, bias, hf, feat16,
                                                                          (long long)N * HCH);
    };

    // layer 1 (input K=32 padded from 26)
    gemm_wmma<<<gemmEt, T, 0, stream>>>(eafull16, We1_16, nullptr, eh, Etot, 32);
    run_layer(feat16, 32, Wl1_16, bl1, Wr1_16, br1, att1, bias1);

    // layers 2 & 3 share the We2-based edge transform
    gemm_wmma<<<gemmEt, T, 0, stream>>>(eafull16, We2_16, nullptr, eh, Etot, 32);
    run_layer(feat16, HCH, Wl2_16, bl2, Wr2_16, br2, att2, bias2);
    run_layer(feat16, HCH, Wl2_16, bl2, Wr2_16, br2, att2, bias2);

    // ---- global max pool ----
    const long long outN = out_size;
    fill_f32<<<blk1(outN), T, 0, stream>>>((float*)d_out, 0.0f, outN);
    pool_kernel<<<(unsigned)((N + 7) / 8), T, 0, stream>>>(hf, batch, (unsigned int*)d_out, N);
}
